// AttentionConvolution2D_20255065768123
// MI455X (gfx1250) — matrix operand order repaired and checked by reading, not compiled, not run on hardware
//
#include <hip/hip_runtime.h>

// MI455X / gfx1250, wave32. Fused blocked-attention kernel:
// one 256-thread workgroup (8 waves) per 4x4 pixel block, everything in LDS,
// all GEMMs via v_wmma_f32_16x16x32_bf16 (f32 accumulate).
// Independent dual accumulator chains interleaved to hide WMMA hazard NOPs.

typedef __attribute__((ext_vector_type(16))) __bf16 v16bf;
typedef __attribute__((ext_vector_type(8)))  __bf16 v8bf;
typedef __attribute__((ext_vector_type(8)))  float  v8f;

// LDS element counts / strides (bf16 elements unless noted)
#define XN_S 136   // xn row stride (128 ch + pad), 272B (16B aligned)
#define QB_S 264   // q/obuf row stride (256 + pad)
#define KB_S 264
#define VT_S 160   // vT row stride (144 keys padded to 160)
#define AB_S 160
#define SMEM_BYTES 263168  // 127488 bf16 + 2048 f32 (pce)

__device__ __forceinline__ v8f wmma_bf16(v16bf a, v16bf b, v8f c) {
  return __builtin_amdgcn_wmma_f32_16x16x32_bf16(false, a, false, b,
                                                 (short)0, c, false, false);
}
// B fragment (KxN, 16-bit): lane n holds column n; elements 0..7 at
// K=kb..kb+7 (kb includes +8 for upper lane half), elements 8..15 at +16.
__device__ __forceinline__ v16bf ldB(const __bf16* p) {
  union { v16bf v; v8bf h[2]; } u;
  u.h[0] = *(const v8bf*)p; u.h[1] = *(const v8bf*)(p + 16); return u.v;
}
// A fragment (16xK, 16-bit): lane holds row (lane&15); elements 0..7 at
// K=kb..kb+7 (kb includes +8 for upper lane half), elements 8..15 at +16.
__device__ __forceinline__ v16bf ldA(const __bf16* p) {
  union { v16bf v; v8bf h[2]; } u;
  u.h[0] = *(const v8bf*)p; u.h[1] = *(const v8bf*)(p + 16); return u.v;
}

__global__ void prep_weights(const float* __restrict__ w_in,
                             const float* __restrict__ w_out,
                             __bf16* __restrict__ ws) {
  int t = blockIdx.x * 256 + threadIdx.x;
  if (t < 98304) ws[t] = (__bf16)w_in[t];           // w_in  [768][128]
  if (t < 32768) ws[98304 + t] = (__bf16)w_out[t];  // w_out [128][256]
}

__global__ __launch_bounds__(256, 1)
void attn_conv2d(const float* __restrict__ x,
                 const float* __restrict__ b_in,
                 const float* __restrict__ b_out,
                 const float* __restrict__ pce,
                 const __bf16* __restrict__ wIn,
                 const __bf16* __restrict__ wOut,
                 float* __restrict__ out) {
  extern __shared__ __align__(16) char smemRaw[];
  __bf16* xn   = (__bf16*)smemRaw;            // [144][XN_S] 12x12 neighborhood (bf16)
  __bf16* qbuf = xn   + 144 * XN_S;           // [16][QB_S]
  __bf16* kbuf = qbuf + 16  * QB_S;           // [144][KB_S]
  __bf16* vT   = kbuf + 144 * KB_S;           // [256][VT_S] v transposed (feat-major)
  __bf16* obuf = vT   + 256 * VT_S;           // [16][QB_S]
  __bf16* abuf = obuf + 16  * QB_S;           // [8][16][AB_S] attn weights per head
  float*  pceL = (float*)(abuf + 8 * 16 * AB_S); // [8][256]

  const int tid = threadIdx.x;
  const int l   = tid & 31;
  const int wv  = tid >> 5;          // wave id 0..7
  const int n   = l & 15;            // lane within half
  const int hi  = (l >> 4) & 1;      // upper lane half
  const int i   = blockIdx.x;        // hb block index (W)
  const int j   = blockIdx.y;        // vb block index (H)
  const int b   = blockIdx.z;
  const int aoff = hi ? 8 : 0;       // A-fragment K offset for upper lane half
  const int boff = hi ? 8 : 0;       // B-fragment K offset for upper lane half

  // ---- Stage A1: cooperative load of 12x12 pixel neighborhood (zero OOB) ----
  const float* xb = x + (size_t)b * 128 * 65536;
  for (int pair = tid; pair < 128 * 12; pair += 256) {
    int c  = pair / 12;
    int h3 = pair - c * 12;                // y3 (H sub-index 0..11)
    int h  = (j << 2) + h3 - 4;
    int w0 = (i << 2) - 4;
    bool hok = (unsigned)h < 256u;
    const float* rowp = xb + (size_t)c * 65536 + (size_t)(hok ? h : 0) * 256;
    #pragma unroll
    for (int t = 0; t < 12; ++t) {         // t = x3 (W sub-index)
      int w = w0 + t;
      float v = 0.0f;
      if (hok && (unsigned)w < 256u) v = rowp[w];
      xn[(t * 12 + h3) * XN_S + c] = (__bf16)v;   // key row kk = x3*12 + y3
    }
  }
  for (int t = tid; t < 2048; t += 256) pceL[t] = pce[t];
  for (int t = tid; t < 256 * 16; t += 256)          // zero key-pad of vT
    vT[(t >> 4) * VT_S + 144 + (t & 15)] = (__bf16)0.0f;
  __syncthreads();

  // ---- Stage A2a: Q projection (center 16 pixels x first 256 cols) ----
  {
    int colL0 = (wv * 2) * 16 + n;
    int colL1 = colL0 + 16;
    const __bf16* w0p = wIn + colL0 * 128 + boff;
    const __bf16* w1p = wIn + colL1 * 128 + boff;
    int arow = 52 + 12 * (n >> 2) + (n & 3);   // center pixel p=n -> xn row
    const __bf16* ap = xn + arow * XN_S + aoff;
    v16bf a0 = ldA(ap), a1 = ldA(ap + 32), a2 = ldA(ap + 64), a3 = ldA(ap + 96);
    v8f q0 = {}, q1 = {};
    q0 = wmma_bf16(a0, ldB(w0p),      q0);
    q1 = wmma_bf16(a0, ldB(w1p),      q1);
    q0 = wmma_bf16(a1, ldB(w0p + 32), q0);
    q1 = wmma_bf16(a1, ldB(w1p + 32), q1);
    q0 = wmma_bf16(a2, ldB(w0p + 64), q0);
    q1 = wmma_bf16(a2, ldB(w1p + 64), q1);
    q0 = wmma_bf16(a3, ldB(w0p + 96), q0);
    q1 = wmma_bf16(a3, ldB(w1p + 96), q1);
    float bi0 = b_in[colL0], bi1 = b_in[colL1];
    #pragma unroll
    for (int r = 0; r < 8; ++r) {
      qbuf[(r + hi * 8) * QB_S + colL0] = (__bf16)(q0[r] + bi0);
      qbuf[(r + hi * 8) * QB_S + colL1] = (__bf16)(q1[r] + bi1);
    }
  }

  // ---- Stage A2b: KV projection (144 keys x 512 cols) ----
  for (int q = 0; q < 4; ++q) {
    int colL = (wv * 4 + q) * 16 + n;          // 0..511 (k: <256, v: >=256)
    const __bf16* wrow = wIn + (256 + colL) * 128 + boff;
    v16bf bF0 = ldB(wrow), bF1 = ldB(wrow + 32), bF2 = ldB(wrow + 64), bF3 = ldB(wrow + 96);
    float bi = b_in[256 + colL];
    bool isK = colL < 256;                     // uniform per tile
    // 4 pairs of M-tiles with independent accumulator chains + 1 tail tile
    for (int mtp = 0; mtp < 4; ++mtp) {
      int mt0 = mtp * 2, mt1 = mt0 + 1;
      const __bf16* ap0 = xn + (mt0 * 16 + n) * XN_S + aoff;
      const __bf16* ap1 = xn + (mt1 * 16 + n) * XN_S + aoff;
      v8f ac0 = {}, ac1 = {};
      ac0 = wmma_bf16(ldA(ap0),      bF0, ac0);
      ac1 = wmma_bf16(ldA(ap1),      bF0, ac1);
      ac0 = wmma_bf16(ldA(ap0 + 32), bF1, ac0);
      ac1 = wmma_bf16(ldA(ap1 + 32), bF1, ac1);
      ac0 = wmma_bf16(ldA(ap0 + 64), bF2, ac0);
      ac1 = wmma_bf16(ldA(ap1 + 64), bF2, ac1);
      ac0 = wmma_bf16(ldA(ap0 + 96), bF3, ac0);
      ac1 = wmma_bf16(ldA(ap1 + 96), bF3, ac1);
      #pragma unroll
      for (int r = 0; r < 8; ++r) {
        int key0 = mt0 * 16 + r + hi * 8;
        int key1 = mt1 * 16 + r + hi * 8;
        float v0 = ac0[r] + bi;
        float v1 = ac1[r] + bi;
        if (isK) {
          kbuf[key0 * KB_S + colL] = (__bf16)v0;
          kbuf[key1 * KB_S + colL] = (__bf16)v1;
        } else {
          vT[(colL - 256) * VT_S + key0] = (__bf16)v0;
          vT[(colL - 256) * VT_S + key1] = (__bf16)v1;
        }
      }
    }
    {
      const __bf16* ap = xn + (8 * 16 + n) * XN_S + aoff;
      v8f acc = {};
      acc = wmma_bf16(ldA(ap),      bF0, acc);
      acc = wmma_bf16(ldA(ap + 32), bF1, acc);
      acc = wmma_bf16(ldA(ap + 64), bF2, acc);
      acc = wmma_bf16(ldA(ap + 96), bF3, acc);
      #pragma unroll
      for (int r = 0; r < 8; ++r) {
        int key = 128 + r + hi * 8;
        float vv = acc[r] + bi;
        if (isK) kbuf[key * KB_S + colL] = (__bf16)vv;
        else     vT[(colL - 256) * VT_S + key] = (__bf16)vv;
      }
    }
  }
  __syncthreads();

  // ---- Stage B: attention, one head per wave ----
  const int hd = wv;
  __bf16* ab = abuf + hd * 16 * AB_S;
  v16bf qA = ldA(qbuf + n * QB_S + hd * 32 + aoff);
  v8f st[9];
  #pragma unroll
  for (int kt = 0; kt < 9; ++kt) {
    v16bf kB = ldB(kbuf + (kt * 16 + n) * KB_S + hd * 32 + boff);
    v8f z = {};
    v8f s = wmma_bf16(qA, kB, z);
    int kk = kt * 16 + n;
    int x3 = kk / 12, y3 = kk - x3 * 12;
    int ii = i + (x3 >> 2) - 1, jj = j + (y3 >> 2) - 1;
    bool oob = ((unsigned)ii >= 64u) || ((unsigned)jj >= 64u);
    int tp = 16 * x3 + y3 + 68;                // 16*(x3+4) + (y3+4)
    #pragma unroll
    for (int r = 0; r < 8; ++r) {
      int p = r + hi * 8;
      int id = tp - (16 * (p >> 2) + (p & 3)); // pce index
      float bias = pceL[hd * 256 + id];
      st[kt][r] = oob ? -1.0e30f : (s[r] * 0.17677669529663687f - bias);
    }
  }
  // softmax across 144 keys: in-register over tiles + shfl_xor within 16-lane half
  float mx[8], sm[8], rs[8];
  #pragma unroll
  for (int r = 0; r < 8; ++r) {
    float m = st[0][r];
    #pragma unroll
    for (int kt = 1; kt < 9; ++kt) m = fmaxf(m, st[kt][r]);
    m = fmaxf(m, __shfl_xor(m, 1, 32));
    m = fmaxf(m, __shfl_xor(m, 2, 32));
    m = fmaxf(m, __shfl_xor(m, 4, 32));
    m = fmaxf(m, __shfl_xor(m, 8, 32));
    mx[r] = m; sm[r] = 0.0f;
  }
  #pragma unroll
  for (int kt = 0; kt < 9; ++kt)
    #pragma unroll
    for (int r = 0; r < 8; ++r) {
      float e = __expf(st[kt][r] - mx[r]);
      st[kt][r] = e; sm[r] += e;
    }
  #pragma unroll
  for (int r = 0; r < 8; ++r) {
    float s2 = sm[r];
    s2 += __shfl_xor(s2, 1, 32);
    s2 += __shfl_xor(s2, 2, 32);
    s2 += __shfl_xor(s2, 4, 32);
    s2 += __shfl_xor(s2, 8, 32);
    rs[r] = 1.0f / s2;
  }
  #pragma unroll
  for (int kt = 0; kt < 9; ++kt)
    #pragma unroll
    for (int r = 0; r < 8; ++r)
      ab[(r + hi * 8) * AB_S + kt * 16 + n] = (__bf16)(st[kt][r] * rs[r]);
  #pragma unroll
  for (int r = 0; r < 8; ++r)                      // zero key-pad cols 144..159
    ab[(r + hi * 8) * AB_S + 144 + n] = (__bf16)0.0f;
  asm volatile("" ::: "memory");                   // order LDS stores before reloads

  // attn(16x160) @ V(160x32): 5 K-chunks x 2 N-tiles (independent chains)
  v8f o0 = {}, o1 = {};
  #pragma unroll
  for (int c = 0; c < 5; ++c) {
    v16bf aF = ldA(ab + n * AB_S + c * 32 + aoff);
    v16bf v0 = ldB(vT + (hd * 32 + n)      * VT_S + c * 32 + boff);
    v16bf v1 = ldB(vT + (hd * 32 + 16 + n) * VT_S + c * 32 + boff);
    o0 = wmma_bf16(aF, v0, o0);
    o1 = wmma_bf16(aF, v1, o1);
  }
  #pragma unroll
  for (int r = 0; r < 8; ++r) {
    int p = r + hi * 8;
    obuf[p * QB_S + hd * 32 + n]      = (__bf16)o0[r];
    obuf[p * QB_S + hd * 32 + 16 + n] = (__bf16)o1[r];
  }
  __syncthreads();

  // ---- Stage C: output projection 256 -> 128, one N-tile per wave ----
  // Two independent 4-deep accumulator chains, summed at the end.
  int co = wv * 16 + n;
  const __bf16* wr = wOut + co * 256 + boff;
  v8f acc0 = {}, acc1 = {};
  #pragma unroll
  for (int k = 0; k < 4; ++k) {
    v16bf aF0 = ldA(obuf + n * QB_S + (2 * k)     * 32 + aoff);
    v16bf aF1 = ldA(obuf + n * QB_S + (2 * k + 1) * 32 + aoff);
    v16bf bF0 = ldB(wr + (2 * k)     * 32);
    v16bf bF1 = ldB(wr + (2 * k + 1) * 32);
    acc0 = wmma_bf16(aF0, bF0, acc0);
    acc1 = wmma_bf16(aF1, bF1, acc1);
  }
  float bo = b_out[co];
  float* op = out + (size_t)(b * 128 + co) * 65536;
  #pragma unroll
  for (int r = 0; r < 8; ++r) {
    int p = r + hi * 8;
    int h = (j << 2) + (p & 3);
    int w = (i << 2) + (p >> 2);
    op[h * 256 + w] = acc0[r] + acc1[r] + bo;
  }
}

extern "C" void kernel_launch(void* const* d_in, const int* in_sizes, int n_in,
                              void* d_out, int out_size, void* d_ws, size_t ws_size,
                              hipStream_t stream) {
  (void)in_sizes; (void)n_in; (void)out_size; (void)ws_size;
  const float* x     = (const float*)d_in[0];
  const float* w_in  = (const float*)d_in[1];
  const float* b_in  = (const float*)d_in[2];
  const float* w_out = (const float*)d_in[3];
  const float* b_out = (const float*)d_in[4];
  const float* pce   = (const float*)d_in[5];
  __bf16* wsb = (__bf16*)d_ws;

  // Allow >64KB dynamic LDS (gfx1250: up to 320KB per workgroup). Idempotent.
  hipFuncSetAttribute((const void*)attn_conv2d,
                      hipFuncAttributeMaxDynamicSharedMemorySize, SMEM_BYTES);

  prep_weights<<<512, 256, 0, stream>>>(w_in, w_out, wsb);
  attn_conv2d<<<dim3(64, 64, 2), 256, SMEM_BYTES, stream>>>(
      x, b_in, b_out, pce, wsb, wsb + 98304, (float*)d_out);
}
